// VectorQuantizer_19181323944129
// MI455X (gfx1250) — matrix operand order repaired and checked by reading, not compiled, not run on hardware
//
#include <hip/hip_runtime.h>

// Problem constants (match reference): B=8, D=512, T=2048, K=8192
#define DDIM   512
#define KCODES 8192
#define TSEQ   2048
#define BATCH  8
#define MTOTAL (BATCH * TSEQ)   // 16384 query vectors
#define MTILE  64               // rows per workgroup (4 row-blocks of 16)
#define NBLK   (MTILE / 16)     // 4
#define NSTEP  128              // 8 waves * 16 columns per sweep step
#define LDS_STRIDE 520          // 512 + 8 bf16 pad -> conflict-free b128 LDS reads

typedef __attribute__((ext_vector_type(16))) __bf16 v16bf;
typedef __attribute__((ext_vector_type(8)))  __bf16 v8bf;
typedef __attribute__((ext_vector_type(8)))  float  v8f;

// ---------------------------------------------------------------------------
// Kernel 1: embedding fp32 -> bf16 hi/lo split, plus per-code squared norm.
// One block per code row. 16 MB read once; bf16 tables stay resident in L2.
// ---------------------------------------------------------------------------
__global__ void vq_prep_codes(const float* __restrict__ emb,
                              __bf16* __restrict__ ehi,
                              __bf16* __restrict__ elo,
                              float* __restrict__ esq) {
    __shared__ float red[256];
    const int k   = blockIdx.x;
    const int tid = threadIdx.x;
    const float* row = emb + (size_t)k * DDIM;
    float s = 0.0f;
    for (int d = tid; d < DDIM; d += 256) {
        float x = row[d];
        __bf16 h = (__bf16)x;                 // RNE -> hi
        __bf16 l = (__bf16)(x - (float)h);    // residual -> lo
        ehi[(size_t)k * DDIM + d] = h;
        elo[(size_t)k * DDIM + d] = l;
        s += x * x;
    }
    red[tid] = s;
    __syncthreads();
    for (int o = 128; o > 0; o >>= 1) {
        if (tid < o) red[tid] += red[tid + o];
        __syncthreads();
    }
    if (tid == 0) esq[k] = red[0];
}

// ---------------------------------------------------------------------------
// Kernel 2: fused GEMM + argmin.
// Workgroup = 256 threads (8 waves), owns 64 rows (4 row-blocks of 16).
// A tile (64 x 512, bf16 hi+lo) staged in LDS once; all 8192 codes streamed.
// Each wave sweeps 16 columns per NSTEP using v_wmma_f32_16x16x32_bf16:
//   cross ~= Ahi*Bhi + Ahi*Blo + Alo*Bhi   (bf16x3, f32 accumulate)
// Each B-fragment pair feeds 4 row-blocks -> 12 WMMAs per B load.
// dist = ||e||^2 - 2*cross  (||x||^2 constant per row -> dropped for argmin)
// ---------------------------------------------------------------------------
__global__ __launch_bounds__(256, 1)
void vq_argmin(const float* __restrict__ latents,
               const __bf16* __restrict__ ehi,
               const __bf16* __restrict__ elo,
               const float* __restrict__ esq,
               int* __restrict__ indices) {
    __shared__ __bf16 Ahi[MTILE * LDS_STRIDE];
    __shared__ __bf16 Alo[MTILE * LDS_STRIDE];
    __shared__ float  rv[8][MTILE];
    __shared__ int    ri[8][MTILE];

    const int tid  = threadIdx.x;
    const int wave = tid >> 5;
    const int lane = tid & 31;
    const int m0   = blockIdx.x * MTILE;     // MTILE divides TSEQ -> single b
    const int b    = m0 / TSEQ;
    const int t0   = m0 % TSEQ;

    // ---- Stage A tile: latents[b, d, t0+tt] -> LDS (row = tt, col = d) ----
    // tid&63 = tt (contiguous t -> coalesced 256B global reads per d-group)
    {
        const float* base = latents + (size_t)b * DDIM * TSEQ + t0;
        const int tt = tid & 63;
        const int dg = tid >> 6;
        for (int d = dg; d < DDIM; d += 4) {
            float x = base[(size_t)d * TSEQ + tt];
            __bf16 h = (__bf16)x;
            __bf16 l = (__bf16)(x - (float)h);
            Ahi[tt * LDS_STRIDE + d] = h;
            Alo[tt * LDS_STRIDE + d] = l;
        }
    }
    __syncthreads();

    const int colLane = lane & 15;   // column within the wave's 16-wide block
    const int half    = lane >> 4;   // selects K-chunk: k = 16*half + s (contiguous order)

    float bestv[NBLK][8];
    int   besti[NBLK][8];
#pragma unroll
    for (int blk = 0; blk < NBLK; ++blk)
#pragma unroll
        for (int v = 0; v < 8; ++v) { bestv[blk][v] = 3.4e38f; besti[blk][v] = 0; }

    const __bf16* ArowHi[NBLK];
    const __bf16* ArowLo[NBLK];
#pragma unroll
    for (int blk = 0; blk < NBLK; ++blk) {
        ArowHi[blk] = Ahi + (size_t)(blk * 16 + colLane) * LDS_STRIDE;
        ArowLo[blk] = Alo + (size_t)(blk * 16 + colLane) * LDS_STRIDE;
    }

    for (int n0 = 0; n0 < KCODES; n0 += NSTEP) {
        const int   mycol = n0 + wave * 16 + colLane;
        const float esq_c = esq[mycol];
        const __bf16* browH = ehi + (size_t)mycol * DDIM;
        const __bf16* browL = elo + (size_t)mycol * DDIM;

        v8f acc[NBLK];
#pragma unroll
        for (int blk = 0; blk < NBLK; ++blk) acc[blk] = (v8f){};

#pragma unroll 2
        for (int d0 = 0; d0 < DDIM; d0 += 32) {
            // B fragment (32x16): lane holds col = colLane, K = d0+half*16 .. +15
            const int bk = d0 + half * 16;
            v16bf bhi, blo;
            ((v8bf*)&bhi)[0] = *(const v8bf*)(browH + bk);
            ((v8bf*)&bhi)[1] = *(const v8bf*)(browH + bk + 8);
            ((v8bf*)&blo)[0] = *(const v8bf*)(browL + bk);
            ((v8bf*)&blo)[1] = *(const v8bf*)(browL + bk + 8);

            // A fragments (16x32): lane holds row = colLane (per block),
            // K = d0+half*16 .. +15
            const int ak = d0 + half * 16;
#pragma unroll
            for (int blk = 0; blk < NBLK; ++blk) {
                v16bf ah, al;
                ((v8bf*)&ah)[0] = *(const v8bf*)(ArowHi[blk] + ak);
                ((v8bf*)&ah)[1] = *(const v8bf*)(ArowHi[blk] + ak + 8);
                ((v8bf*)&al)[0] = *(const v8bf*)(ArowLo[blk] + ak);
                ((v8bf*)&al)[1] = *(const v8bf*)(ArowLo[blk] + ak + 8);

                acc[blk] = __builtin_amdgcn_wmma_f32_16x16x32_bf16(false, ah, false, bhi, (short)0, acc[blk], false, false);
                acc[blk] = __builtin_amdgcn_wmma_f32_16x16x32_bf16(false, ah, false, blo, (short)0, acc[blk], false, false);
                acc[blk] = __builtin_amdgcn_wmma_f32_16x16x32_bf16(false, al, false, bhi, (short)0, acc[blk], false, false);
            }
        }

        // C layout: VGPR v -> row v + 8*half, col = colLane. Same col for all v.
#pragma unroll
        for (int blk = 0; blk < NBLK; ++blk) {
#pragma unroll
            for (int v = 0; v < 8; ++v) {
                float dv = esq_c - 2.0f * acc[blk][v];
                if (dv < bestv[blk][v]) { bestv[blk][v] = dv; besti[blk][v] = mycol; }
            }
        }
    }

    // ---- cross-lane argmin within each 16-lane half (cols live across lanes)
#pragma unroll
    for (int blk = 0; blk < NBLK; ++blk) {
#pragma unroll
        for (int v = 0; v < 8; ++v) {
            float val = bestv[blk][v];
            int   idx = besti[blk][v];
            for (int off = 8; off >= 1; off >>= 1) {
                float ov = __shfl_xor(val, off, 32);
                int   oi = __shfl_xor(idx, off, 32);
                if (ov < val || (ov == val && oi < idx)) { val = ov; idx = oi; }
            }
            if (colLane == 0) {
                const int rowl = blk * 16 + half * 8 + v;
                rv[wave][rowl] = val;
                ri[wave][rowl] = idx;
            }
        }
    }
    __syncthreads();

    // ---- combine the 8 waves' candidates per row, write final index ----
    if (tid < MTILE) {
        float val = rv[0][tid];
        int   idx = ri[0][tid];
#pragma unroll
        for (int w = 1; w < 8; ++w) {
            float ov = rv[w][tid];
            int   oi = ri[w][tid];
            if (ov < val || (ov == val && oi < idx)) { val = ov; idx = oi; }
        }
        indices[m0 + tid] = idx;
    }
}

// ---------------------------------------------------------------------------
// Kernel 3: gather + transpose: out[b, d, t] = emb[idx[b*T+t]][d]
// 64 contiguous t's per block -> 256B contiguous stores per d.
// ---------------------------------------------------------------------------
__global__ void vq_gather(const float* __restrict__ emb,
                          const int* __restrict__ indices,
                          float* __restrict__ out) {
    const int m0 = blockIdx.x * 64;
    const int b  = m0 / TSEQ;
    const int t0 = m0 % TSEQ;
    const int tl = threadIdx.x & 63;
    const int q  = threadIdx.x >> 6;
    const int code = indices[m0 + tl];
    const float* erow = emb + (size_t)code * DDIM;
    float* obase = out + (size_t)b * DDIM * TSEQ + t0 + tl;
    for (int d = q; d < DDIM; d += 4) {
        obase[(size_t)d * TSEQ] = erow[d];
    }
}

// ---------------------------------------------------------------------------
extern "C" void kernel_launch(void* const* d_in, const int* in_sizes, int n_in,
                              void* d_out, int out_size, void* d_ws, size_t ws_size,
                              hipStream_t stream) {
    const float* latents = (const float*)d_in[0];  // [B, D, T] fp32
    const float* emb     = (const float*)d_in[1];  // [K, D] fp32
    float*       out     = (float*)d_out;          // [B, D, T] fp32

    // Workspace layout: ehi (8MB) | elo (8MB) | esq (32KB) | indices (64KB)
    char*   ws  = (char*)d_ws;
    __bf16* ehi = (__bf16*)ws;
    __bf16* elo = (__bf16*)(ws + (size_t)KCODES * DDIM * 2);
    float*  esq = (float*)(ws + (size_t)KCODES * DDIM * 4);
    int*    idx = (int*)(ws + (size_t)KCODES * DDIM * 4 + (size_t)KCODES * 4);

    vq_prep_codes<<<KCODES, 256, 0, stream>>>(emb, ehi, elo, esq);
    vq_argmin<<<MTOTAL / MTILE, 256, 0, stream>>>(latents, ehi, elo, esq, idx);
    vq_gather<<<MTOTAL / 64, 256, 0, stream>>>(emb, idx, out);
}
